// VisionReasoningNet_3118146257370
// MI455X (gfx1250) — compile-verified, matrix operand order checked by reading, not run on hardware
//
#include <hip/hip_runtime.h>
#include <hip/hip_bf16.h>
#include <math.h>

#define NUM_ENT 100000
#define NUM_REL 500
#define BSZ     16
#define T_TRIP  4096
#define QLEN    32
#define KC      4
#define DIM     768
#define STEPS   3

typedef __bf16 bf16_t;
typedef __attribute__((ext_vector_type(16))) __bf16 v16bf;
typedef __attribute__((ext_vector_type(8)))  float  v8f;

// POD 128-bit / 256-bit helpers (guaranteed trivially copyable for bit_cast)
struct U128 { unsigned int a, b, c, d; };
struct Frag256 { U128 lo, hi; };            // 32 bytes == v16bf
struct F4 { float x, y, z, w; };

__device__ __forceinline__ float wave_sum32(float v) {
#pragma unroll
  for (int o = 16; o > 0; o >>= 1) v += __shfl_down(v, o, 32);
  return v;
}
__device__ __forceinline__ float sigmoidf_(float x) { return 1.0f / (1.0f + expf(-x)); }

// ---------------- utility kernels ----------------

__global__ void zero_kernel(float* __restrict__ p, size_t n) {
  for (size_t i = (size_t)blockIdx.x * blockDim.x + threadIdx.x; i < n;
       i += (size_t)gridDim.x * blockDim.x)
    p[i] = 0.0f;
}

__device__ __forceinline__ unsigned short f2bf_rne(float f) {
  unsigned u = __float_as_uint(f);
  return (unsigned short)((u + 0x7FFFu + ((u >> 16) & 1u)) >> 16);
}

__global__ void f2bf_kernel(const float* __restrict__ in, unsigned short* __restrict__ out, int n) {
  for (int i = blockIdx.x * blockDim.x + threadIdx.x; i < n; i += gridDim.x * blockDim.x)
    out[i] = f2bf_rne(in[i]);
}

// step_W [STEPS][k][n] (f32) -> Wt_bf [STEPS][n][k] (bf16): contiguous-K rows for WMMA B frags
__global__ void f2bf_transpose_kernel(const float* __restrict__ in, unsigned short* __restrict__ out) {
  const int n_tot = STEPS * DIM * DIM;
  for (int i = blockIdx.x * blockDim.x + threadIdx.x; i < n_tot; i += gridDim.x * blockDim.x) {
    int k    = i % DIM;
    int n    = (i / DIM) % DIM;
    int step = i / (DIM * DIM);
    out[i] = f2bf_rne(in[(size_t)step * DIM * DIM + (size_t)k * DIM + n]);
  }
}

// scatter 1/K at keyconcepts -> laste0 (already L1-normalized since sum==K)
__global__ void seed_kernel(const int* __restrict__ kc, float* __restrict__ laste0, float val) {
  int i = threadIdx.x;
  if (i < BSZ * KC) {
    int b = i / KC;
    atomicAdd(&laste0[(size_t)b * NUM_ENT + kc[i]], val);
  }
}

// ---------------- cq = tanh(text_emb @ step_W[t] + step_b[t])  (WMMA) ----------------
// one wave per 16x16 output tile; grid = (DIM/16, STEPS)
// A row-major [M][K], B pre-transposed [N][K] -> both frag halves are contiguous b128 loads.

__global__ __launch_bounds__(32)
void cq_gemm_kernel(const unsigned short* __restrict__ A_bf,   // [BSZ][DIM] bf16
                    const unsigned short* __restrict__ Wt_bf,  // [STEPS][DIM(n)][DIM(k)] bf16
                    const float* __restrict__ step_b,          // [STEPS][DIM]
                    float* __restrict__ cq)                    // [STEPS][BSZ][DIM]
{
  const int step = blockIdx.y;
  const int n0   = blockIdx.x * 16;
  const int lane = threadIdx.x;
  const int half = lane >> 4;     // K-half select per ISA 16-bit A/B fragment layout
  const int l15  = lane & 15;
  const unsigned short* arow = A_bf + (size_t)l15 * DIM;                         // A row M=l15
  const unsigned short* brow = Wt_bf + ((size_t)step * DIM + n0 + l15) * DIM;    // B col N=n0+l15

  v8f acc = {};
  for (int k0 = 0; k0 < DIM; k0 += 32) {
    const int kb = k0 + half * 8;  // lanes 0-15: K k0..k0+7 / +16..23 ; lanes 16-31: +8.. / +24..
    Frag256 fa, fb;
    fa.lo = *(const U128*)(arow + kb);
    fa.hi = *(const U128*)(arow + kb + 16);
    fb.lo = *(const U128*)(brow + kb);
    fb.hi = *(const U128*)(brow + kb + 16);
    v16bf a = __builtin_bit_cast(v16bf, fa);
    v16bf b = __builtin_bit_cast(v16bf, fb);
    acc = __builtin_amdgcn_wmma_f32_16x16x32_bf16(false, a, false, b, (short)0, acc,
                                                  false, false);
  }
#pragma unroll
  for (int j = 0; j < 8; ++j) {
    int row = j + half * 8;        // C layout: VGPR j -> M=j (lanes 0-15) / M=j+8 (lanes 16-31)
    int col = n0 + l15;
    cq[((size_t)step * BSZ + row) * DIM + col] = tanhf(acc[j] + step_b[step * DIM + col]);
  }
}

// ---------------- question attention -> ctx ; precompute ctx*w_rel, ctx*w_ent, type_s ----

__global__ __launch_bounds__(256)
void ctx_kernel(const float* __restrict__ q_word,  // [BSZ][QLEN][DIM]
                const float* __restrict__ cq,      // [STEPS][BSZ][DIM]
                const float* __restrict__ w_rel, const float* __restrict__ w_ent,
                const float* __restrict__ w_type, const float* __restrict__ b_type,
                float* __restrict__ ctxw_rel,      // [STEPS][BSZ][DIM]
                float* __restrict__ ctxw_ent,      // [STEPS][BSZ][DIM]
                float* __restrict__ type_s)        // [STEPS][BSZ]
{
  __shared__ float s_logit[QLEN];
  __shared__ float s_dist[QLEN];
  __shared__ float s_red[256];
  const int b = blockIdx.x, step = blockIdx.y;
  const int tid = threadIdx.x, lane = tid & 31, wave = tid >> 5;
  const float* cqv = cq + ((size_t)step * BSZ + b) * DIM;
  const float* qw  = q_word + (size_t)b * QLEN * DIM;

  for (int q = wave; q < QLEN; q += 8) {
    const float* row = qw + (size_t)q * DIM;
    float p = 0.0f;
    for (int d = lane; d < DIM; d += 32) p = fmaf(cqv[d], row[d], p);
    p = wave_sum32(p);
    if (lane == 0) s_logit[q] = p;
  }
  __syncthreads();
  if (tid == 0) {
    float m = -1e30f;
    for (int q = 0; q < QLEN; ++q) m = fmaxf(m, s_logit[q]);
    float s = 0.0f;
    for (int q = 0; q < QLEN; ++q) { float e = expf(s_logit[q] - m); s_dist[q] = e; s += e; }
    float inv = 1.0f / s, s2 = 0.0f;
    for (int q = 0; q < QLEN; ++q) { s_dist[q] *= inv; s2 += s_dist[q]; }
    float inv2 = 1.0f / (s2 + 1e-6f);
    for (int q = 0; q < QLEN; ++q) s_dist[q] *= inv2;
  }
  __syncthreads();

  float tdot = 0.0f;
  for (int d = tid; d < DIM; d += 256) {
    float acc = cqv[d];
#pragma unroll 8
    for (int q = 0; q < QLEN; ++q) acc = fmaf(s_dist[q], qw[(size_t)q * DIM + d], acc);
    size_t o = ((size_t)step * BSZ + b) * DIM + d;
    ctxw_rel[o] = acc * w_rel[d];
    ctxw_ent[o] = acc * w_ent[d];
    tdot = fmaf(acc, w_type[d], tdot);
  }
  s_red[tid] = tdot; __syncthreads();
  for (int k = 128; k > 0; k >>= 1) { if (tid < k) s_red[tid] += s_red[tid + k]; __syncthreads(); }
  if (tid == 0) type_s[step * BSZ + b] = sigmoidf_(s_red[0] + b_type[0]);
}

// ---------------- dominant kernel: gated triple scoring + scatter ----------------
// one wave32 per triple; inactive triples exit after a 4-byte sub_score probe.
// Active path: 6 iterations of coalesced b128 loads per table row (float4).

__global__ __launch_bounds__(256)
void triple_kernel(const float* __restrict__ laste,     // [BSZ][NUM_ENT] normalized prev
                   const float* __restrict__ ctxw_rel,  // [BSZ][DIM] (this step)
                   const float* __restrict__ ctxw_ent,  // [BSZ][DIM]
                   const float* __restrict__ type_s,    // [BSZ]
                   const float* __restrict__ rel_table, // [NUM_REL][DIM]
                   const float* __restrict__ ent_table, // [NUM_ENT][DIM]
                   const int* __restrict__ sub, const int* __restrict__ rel,
                   const int* __restrict__ obj,
                   const float* __restrict__ b_rel_p, const float* __restrict__ b_ent_p,
                   float margin,
                   float* __restrict__ ent_acc,         // [BSZ][NUM_ENT]
                   float* __restrict__ rel_acc)         // [BSZ][NUM_REL]
{
  const int wid  = (blockIdx.x * 256 + threadIdx.x) >> 5;
  const int lane = threadIdx.x & 31;
  const int b = wid >> 12;          // T_TRIP = 4096
  const int t = wid & (T_TRIP - 1);
  const int it = b * T_TRIP + t;

  const int   s  = sub[it];
  const float ss = laste[(size_t)b * NUM_ENT + s];
  if (ss <= margin) return;         // wave-uniform gate: skip gathers entirely

  const int r = rel[it];
  const int o = obj[it];
  const F4* cr = (const F4*)(ctxw_rel + (size_t)b * DIM);
  const F4* ce = (const F4*)(ctxw_ent + (size_t)b * DIM);
  const F4* rr = (const F4*)(rel_table + (size_t)r * DIM);
  const F4* er = (const F4*)(ent_table + (size_t)o * DIM);

  float dr = 0.0f, de = 0.0f;
#pragma unroll
  for (int i = 0; i < DIM / 128; ++i) {   // 6 iters, 32 lanes x float4
    const int idx = i * 32 + lane;
    F4 c = cr[idx], v = rr[idx];
    dr = fmaf(c.x, v.x, dr); dr = fmaf(c.y, v.y, dr);
    dr = fmaf(c.z, v.z, dr); dr = fmaf(c.w, v.w, dr);
    F4 c2 = ce[idx], v2 = er[idx];
    de = fmaf(c2.x, v2.x, de); de = fmaf(c2.y, v2.y, de);
    de = fmaf(c2.z, v2.z, de); de = fmaf(c2.w, v2.w, de);
  }
  dr = wave_sum32(dr);
  de = wave_sum32(de);
  if (lane == 0) {
    float rp = sigmoidf_(dr + b_rel_p[0]);
    float ep = sigmoidf_(de + b_ent_p[0]);
    float ts = type_s[b];
    float tp = ts * rp + (1.0f - ts) * ep;
    atomicAdd(&rel_acc[(size_t)b * NUM_REL + r], tp);
    atomicAdd(&ent_acc[(size_t)b * NUM_ENT + o], ss * tp);
  }
}

// ---------------- per-batch L1 normalization (in place) ----------------

__global__ __launch_bounds__(256)
void l1norm_kernel(float* __restrict__ p, int n) {
  __shared__ float red[256];
  __shared__ float s_inv;
  float* row = p + (size_t)blockIdx.x * n;
  const int tid = threadIdx.x;
  float s = 0.0f;
  for (int i = tid; i < n; i += 256) s += fabsf(row[i]);
  red[tid] = s; __syncthreads();
  for (int k = 128; k > 0; k >>= 1) { if (tid < k) red[tid] += red[tid + k]; __syncthreads(); }
  if (tid == 0) s_inv = 1.0f / fmaxf(red[0], 1e-12f);
  __syncthreads();
  const float inv = s_inv;
  for (int i = tid; i < n; i += 256) row[i] *= inv;
}

// ---------------- hop / answer attention ----------------

__global__ __launch_bounds__(128)
void hopans_kernel(const float* __restrict__ te, const float* __restrict__ W_hop,
                   const float* __restrict__ b_hop, const float* __restrict__ W_ans,
                   const float* __restrict__ b_ans, float* __restrict__ hop_attn,
                   float* __restrict__ ansp) {
  __shared__ float red[128];
  __shared__ float logit[5];
  const int b = blockIdx.x, tid = threadIdx.x;
  const float* x = te + (size_t)b * DIM;
  for (int j = 0; j < 5; ++j) {
    float p = 0.0f;
    if (j < 3) { for (int d = tid; d < DIM; d += 128) p = fmaf(x[d], W_hop[d * 3 + j], p); }
    else       { int jj = j - 3;
                 for (int d = tid; d < DIM; d += 128) p = fmaf(x[d], W_ans[d * 2 + jj], p); }
    red[tid] = p; __syncthreads();
    for (int k = 64; k > 0; k >>= 1) { if (tid < k) red[tid] += red[tid + k]; __syncthreads(); }
    if (tid == 0) logit[j] = red[0] + (j < 3 ? b_hop[j] : b_ans[j - 3]);
    __syncthreads();
  }
  if (tid == 0) {
    float m = fmaxf(fmaxf(logit[0], logit[1]), logit[2]);
    float e0 = expf(logit[0] - m), e1 = expf(logit[1] - m), e2 = expf(logit[2] - m);
    float s = e0 + e1 + e2;
    hop_attn[b * 3 + 0] = e0 / s; hop_attn[b * 3 + 1] = e1 / s; hop_attn[b * 3 + 2] = e2 / s;
    float ma = fmaxf(logit[3], logit[4]);
    float a0 = expf(logit[3] - ma), a1 = expf(logit[4] - ma);
    ansp[b * 2 + 0] = a0 / (a0 + a1); ansp[b * 2 + 1] = a1 / (a0 + a1);
  }
}

// ---------------- final combine -> pred [BSZ][NUM_ENT+NUM_REL] ----------------

__global__ void combine_kernel(const float* __restrict__ ent_prob,  // [STEPS][BSZ][NUM_ENT]
                               const float* __restrict__ rel_prob,  // [STEPS][BSZ][NUM_REL]
                               const float* __restrict__ hop_attn,  // [BSZ][3]
                               const float* __restrict__ ansp,      // [BSZ][2]
                               float* __restrict__ out) {
  const int OUTW = NUM_ENT + NUM_REL;
  const size_t n = (size_t)BSZ * OUTW;
  for (size_t i = (size_t)blockIdx.x * blockDim.x + threadIdx.x; i < n;
       i += (size_t)gridDim.x * blockDim.x) {
    const int b = (int)(i / OUTW);
    const int j = (int)(i % OUTW);
    const float h0 = hop_attn[b * 3 + 0], h1 = hop_attn[b * 3 + 1], h2 = hop_attn[b * 3 + 2];
    float v;
    if (j < NUM_ENT) {
      const size_t base = (size_t)b * NUM_ENT + j;
      const size_t stp  = (size_t)BSZ * NUM_ENT;
      v = ansp[b * 2 + 0] *
          (h0 * ent_prob[base] + h1 * ent_prob[base + stp] + h2 * ent_prob[base + 2 * stp]);
    } else {
      const size_t base = (size_t)b * NUM_REL + (j - NUM_ENT);
      const size_t stp  = (size_t)BSZ * NUM_REL;
      v = ansp[b * 2 + 1] *
          (h0 * rel_prob[base] + h1 * rel_prob[base + stp] + h2 * rel_prob[base + 2 * stp]);
    }
    out[i] = v;
  }
}

// ---------------- launch ----------------

extern "C" void kernel_launch(void* const* d_in, const int* in_sizes, int n_in,
                              void* d_out, int out_size, void* d_ws, size_t ws_size,
                              hipStream_t stream) {
  const float* text_emb  = (const float*)d_in[0];
  const float* q_word    = (const float*)d_in[1];
  const float* ent_table = (const float*)d_in[2];
  const float* rel_table = (const float*)d_in[3];
  const float* step_W    = (const float*)d_in[4];
  const float* step_b    = (const float*)d_in[5];
  const float* w_rel     = (const float*)d_in[6];
  const float* b_rel     = (const float*)d_in[7];
  const float* w_ent     = (const float*)d_in[8];
  const float* b_ent     = (const float*)d_in[9];
  const float* w_type    = (const float*)d_in[10];
  const float* b_type    = (const float*)d_in[11];
  const float* W_hop     = (const float*)d_in[12];
  const float* b_hop     = (const float*)d_in[13];
  const float* W_ans     = (const float*)d_in[14];
  const float* b_ans     = (const float*)d_in[15];
  const int*   keycon    = (const int*)d_in[16];
  const int*   sub       = (const int*)d_in[17];
  const int*   rel       = (const int*)d_in[18];
  const int*   obj       = (const int*)d_in[19];
  float* out = (float*)d_out;

  // workspace carve-out (~30 MB total)
  float* ws        = (float*)d_ws;
  float* laste0    = ws;                                           // BSZ*NUM_ENT
  float* ent_prob  = laste0   + (size_t)BSZ * NUM_ENT;             // STEPS*BSZ*NUM_ENT
  float* rel_prob  = ent_prob + (size_t)STEPS * BSZ * NUM_ENT;     // STEPS*BSZ*NUM_REL
  float* cq        = rel_prob + (size_t)STEPS * BSZ * NUM_REL;     // STEPS*BSZ*DIM
  float* ctxw_rel  = cq       + (size_t)STEPS * BSZ * DIM;
  float* ctxw_ent  = ctxw_rel + (size_t)STEPS * BSZ * DIM;
  float* type_s    = ctxw_ent + (size_t)STEPS * BSZ * DIM;         // STEPS*BSZ
  float* hop_attn  = type_s   + STEPS * BSZ;                       // BSZ*3
  float* ansp      = hop_attn + BSZ * STEPS;                       // BSZ*2
  unsigned short* te_bf = (unsigned short*)(ansp + BSZ * 2);       // BSZ*DIM
  unsigned short* Wt_bf = te_bf + (size_t)BSZ * DIM;               // STEPS*DIM*DIM (transposed)

  const float margin = 1.0f / (float)KC - 1e-6f;  // z == KC exactly (sum of K ones)

  // 1) zero accumulators (laste0 + ent_prob + rel_prob) every call
  const size_t n_zero = (size_t)BSZ * NUM_ENT * (1 + STEPS) + (size_t)STEPS * BSZ * NUM_REL;
  zero_kernel<<<2048, 256, 0, stream>>>(ws, n_zero);

  // 2) bf16 copies for WMMA GEMM (W pre-transposed so B frags are contiguous b128 loads)
  f2bf_kernel<<<48, 256, 0, stream>>>(text_emb, te_bf, BSZ * DIM);
  f2bf_transpose_kernel<<<2048, 256, 0, stream>>>(step_W, Wt_bf);

  // 3) seed initial entity distribution (scatter 1/K == L1-normalized)
  seed_kernel<<<1, 64, 0, stream>>>(keycon, laste0, 1.0f / (float)KC);

  // 4) cq for all steps (WMMA), then ctx / gate precompute for all steps
  cq_gemm_kernel<<<dim3(DIM / 16, STEPS), 32, 0, stream>>>(te_bf, Wt_bf, step_b, cq);
  ctx_kernel<<<dim3(BSZ, STEPS), 256, 0, stream>>>(q_word, cq, w_rel, w_ent, w_type, b_type,
                                                   ctxw_rel, ctxw_ent, type_s);

  // 5) hop / answer attention (independent of steps)
  hopans_kernel<<<BSZ, 128, 0, stream>>>(text_emb, W_hop, b_hop, W_ans, b_ans, hop_attn, ansp);

  // 6) sequential hops: gated triple scoring + scatter, then L1 norms
  for (int s = 0; s < STEPS; ++s) {
    const float* laste = (s == 0) ? laste0 : (ent_prob + (size_t)(s - 1) * BSZ * NUM_ENT);
    float* eacc = ent_prob + (size_t)s * BSZ * NUM_ENT;
    float* racc = rel_prob + (size_t)s * BSZ * NUM_REL;
    triple_kernel<<<(BSZ * T_TRIP * 32) / 256, 256, 0, stream>>>(
        laste, ctxw_rel + (size_t)s * BSZ * DIM, ctxw_ent + (size_t)s * BSZ * DIM,
        type_s + s * BSZ, rel_table, ent_table, sub, rel, obj, b_rel, b_ent, margin,
        eacc, racc);
    l1norm_kernel<<<BSZ, 256, 0, stream>>>(eacc, NUM_ENT);
    l1norm_kernel<<<BSZ, 256, 0, stream>>>(racc, NUM_REL);
  }

  // 7) final combine
  combine_kernel<<<6282, 256, 0, stream>>>(ent_prob, rel_prob, hop_attn, ansp, out);
}
